// _NodeEdgeBlock_89721866813835
// MI455X (gfx1250) — matrix operand order repaired and checked by reading, not compiled, not run on hardware
//
#include <hip/hip_runtime.h>

typedef __bf16 bf16;
typedef __attribute__((ext_vector_type(16))) __bf16 v16bf;
typedef __attribute__((ext_vector_type(8)))  float  v8f;
typedef __attribute__((ext_vector_type(4)))  int    v4i;

#define BS 2
#define NN 256
#define DX 256
#define DE 64
#define DY 64
#define NH 8
#define DF 32

// 1/sqrt(DF)
#define RS 0.17677669529663687f

#if defined(__has_builtin)
#if __has_builtin(__builtin_amdgcn_global_load_async_to_lds_b128) && \
    __has_builtin(__builtin_amdgcn_s_wait_asynccnt)
#define HAVE_ASYNC_LDS 1
#endif
#endif

#if defined(HAVE_ASYNC_LDS)
typedef __attribute__((address_space(1))) v4i* gptr4;
typedef __attribute__((address_space(3))) v4i* lptr4;
#endif

// ---- WMMA fragment index maps (CDNA5 ISA 7.12.2, wave32) ----
// A (16x32 bf16): lane&15 = row M, g=lane>>4; element e -> K = (e&7) + 8g + 16*(e>>3)
__device__ __forceinline__ int amap(int e, int g) { return (e & 7) + (g << 3) + ((e >> 3) << 4); }
// B (32x16 bf16): lane&15 = col N, g=lane>>4; element e -> K = (e&7) + 8g + 16*(e>>3)
__device__ __forceinline__ int bmap(int e, int g) { return (e & 7) + (g << 3) + ((e >> 3) << 4); }
// C/D (16x16 f32): element v -> row M = v + 8g, col N = lane&15

__device__ __forceinline__ v8f wmma_bf16(v16bf a, v16bf b, v8f c) {
  return __builtin_amdgcn_wmma_f32_16x16x32_bf16(false, a, false, b, (short)0, c, false, false);
}

__device__ __forceinline__ v8f vzero8() {
  v8f z;
#pragma unroll
  for (int i = 0; i < 8; ++i) z[i] = 0.f;
  return z;
}

// A-fragment from a row-major f32 row (one 32-wide K slice starting at `row`).
// K runs are contiguous: e<8 -> K = 8g+e ; e>=8 -> K = 16+8g+(e-8). Use float4 loads.
__device__ __forceinline__ v16bf load_a_frag_f32(const float* __restrict__ row, int g) {
  const float4* p0 = (const float4*)(row + (g << 3));
  const float4* p1 = (const float4*)(row + 16 + (g << 3));
  float4 x0 = p0[0], x1 = p0[1], x2 = p1[0], x3 = p1[1];
  v16bf a;
  a[0] = (bf16)x0.x;  a[1] = (bf16)x0.y;  a[2] = (bf16)x0.z;  a[3] = (bf16)x0.w;
  a[4] = (bf16)x1.x;  a[5] = (bf16)x1.y;  a[6] = (bf16)x1.z;  a[7] = (bf16)x1.w;
  a[8] = (bf16)x2.x;  a[9] = (bf16)x2.y;  a[10] = (bf16)x2.z; a[11] = (bf16)x2.w;
  a[12] = (bf16)x3.x; a[13] = (bf16)x3.y; a[14] = (bf16)x3.z; a[15] = (bf16)x3.w;
  return a;
}

// ------------------------------------------------------------------
// Kernel 1: convert weights to bf16 workspace copies.
// Wqkv: [256][768] = q|k|v concatenated along output columns.
// Wem|Wea|Weo are CONTIGUOUS in ws (96 KB) so k_fused can stage them in one shot.
// ------------------------------------------------------------------
__global__ __launch_bounds__(256) void k_cvt_weights(
    const float* __restrict__ qw, const float* __restrict__ kw, const float* __restrict__ vw,
    const float* __restrict__ emw, const float* __restrict__ eaw, const float* __restrict__ eow,
    bf16* __restrict__ Wqkv, bf16* __restrict__ Wem, bf16* __restrict__ Wea, bf16* __restrict__ Weo) {
  int idx = blockIdx.x * 256 + threadIdx.x;
  if (idx < 196608) {
    int k = idx / 768, c = idx % 768;
    float v = (c < 256) ? qw[k * 256 + c] : (c < 512) ? kw[k * 256 + c - 256] : vw[k * 256 + c - 512];
    Wqkv[idx] = (bf16)v;
  } else if (idx < 212992) {
    int r = idx - 196608; Wem[r] = (bf16)emw[r];
  } else if (idx < 229376) {
    int r = idx - 212992; Wea[r] = (bf16)eaw[r];
  } else if (idx < 245760) {
    int r = idx - 229376; Weo[r] = (bf16)eow[r];
  }
}

// ------------------------------------------------------------------
// Kernel 2: fused QKV projection. QKV[row][0:256]=Q, [256:512]=K, [512:768]=V
// C = (X @ Wqkv + bias) * xm[row].  M=512, K=256, N=768.
// Block: 8 waves -> 128(M) x 64(N) tile. Grid (4, 12).
// ------------------------------------------------------------------
__global__ __launch_bounds__(256) void k_qkv_gemm(
    const float* __restrict__ X, const bf16* __restrict__ Wqkv,
    const float* __restrict__ qb, const float* __restrict__ kb, const float* __restrict__ vb,
    const unsigned char* __restrict__ mask, float* __restrict__ QKV) {
  int lane = threadIdx.x & 31, w = threadIdx.x >> 5;
  int g = lane >> 4, mr = lane & 15, col = lane & 15;
  int m0 = blockIdx.x * 128 + w * 16;
  int n0 = blockIdx.y * 64;

  v8f acc[4];
#pragma unroll
  for (int nt = 0; nt < 4; ++nt) acc[nt] = vzero8();

  for (int kk = 0; kk < 256; kk += 32) {
    v16bf a = load_a_frag_f32(X + (size_t)(m0 + mr) * 256 + kk, g);
#pragma unroll
    for (int nt = 0; nt < 4; ++nt) {
      v16bf bb;
      int c = n0 + nt * 16 + col;
#pragma unroll
      for (int e = 0; e < 16; ++e) bb[e] = Wqkv[(size_t)(kk + bmap(e, g)) * 768 + c];
      acc[nt] = wmma_bf16(a, bb, acc[nt]);
    }
  }
#pragma unroll
  for (int nt = 0; nt < 4; ++nt) {
    int c = n0 + nt * 16 + col;
    float bias = (c < 256) ? qb[c] : (c < 512) ? kb[c - 256] : vb[c - 512];
#pragma unroll
    for (int v = 0; v < 8; ++v) {
      int row = m0 + v + 8 * g;
      float m = mask[row] ? 1.f : 0.f;
      QKV[(size_t)row * 768 + c] = (acc[nt][v] + bias) * m;
    }
  }
}

// ------------------------------------------------------------------
// Kernel 3: small y-conditioned vectors.
// yev layout (floats): [0:512]=ye1 (y@y_e_add), [512:1024]=ye2 (y@y_e_mul),
//                      [1024:1536]=yx1 (y@y_x_add), [1536:2048]=yx2 (y@y_x_mul)
// ------------------------------------------------------------------
__global__ __launch_bounds__(256) void k_yvec(
    const float* __restrict__ y,
    const float* __restrict__ w_yea, const float* __restrict__ b_yea,
    const float* __restrict__ w_yem, const float* __restrict__ b_yem,
    const float* __restrict__ w_yxa, const float* __restrict__ b_yxa,
    const float* __restrict__ w_yxm, const float* __restrict__ b_yxm,
    float* __restrict__ yev) {
  int idx = blockIdx.x * 256 + threadIdx.x;  // 0..2047
  int which = idx >> 9, rem = idx & 511, b = rem >> 8, c = rem & 255;
  const float* W; const float* B;
  if (which == 0)      { W = w_yea; B = b_yea; }
  else if (which == 1) { W = w_yem; B = b_yem; }
  else if (which == 2) { W = w_yxa; B = b_yxa; }
  else                 { W = w_yxm; B = b_yxm; }
  float acc = B[c];
  for (int k = 0; k < 64; ++k) acc += y[b * 64 + k] * W[k * 256 + c];
  yev[idx] = acc;  // idx == which*512 + b*256 + c
}

// ------------------------------------------------------------------
// Kernel 4: fused per-(b,i) node/edge block.
//  Stage 0: async-stage Wem|Wea|Weo (96KB bf16, contiguous) into LDS
//  Stage 1: Y[j,c] = Q[i,c]*K[j,c]/sqrt(DF) * (E1+1) + E2 via WMMA -> LDS bf16 slab
//  Stage 2: e_out GEMM: newE = ((ye1 + (ye2+1)*Y) @ Weo + b) * mask  via WMMA
//  Stage 3: per-channel softmax over j + weighted V -> tmpv
//  Stage 4: newX row = (tmpv @ x_out_w + b) * xm[i]
// Dynamic LDS: 98304 (weights) + 131072 (Ybf) + 4*1024 = 233472 B  (WGP has 320KB)
// ------------------------------------------------------------------
#define FUSED_LDS_BYTES 233472

__global__ __launch_bounds__(256) void k_fused(
    const float* __restrict__ E, const unsigned char* __restrict__ mask,
    const float* __restrict__ QKV,
    const bf16* __restrict__ Wem,  // Wem|Wea|Weo contiguous, 98304 bytes total
    const float* __restrict__ emul_b, const float* __restrict__ eadd_b, const float* __restrict__ eout_b,
    const float* __restrict__ yev,
    const float* __restrict__ x_out_w, const float* __restrict__ x_out_b,
    float* __restrict__ outX, float* __restrict__ outE) {
  extern __shared__ char smem[];
  bf16* sWem = (bf16*)smem;                      // 64*256 bf16
  bf16* sWea = (bf16*)(smem + 32768);            // 64*256 bf16
  bf16* sWeo = (bf16*)(smem + 65536);            // 256*64 bf16
  bf16* Ybf  = (bf16*)(smem + 98304);            // 256*256 bf16
  float* qrow = (float*)(smem + 229376);         // 256
  float* ye1s = qrow + 256;                      // 256
  float* ye2s = ye1s + 256;                      // 256
  float* tmpv = ye2s + 256;                      // 256

  int b = blockIdx.x >> 8, i = blockIdx.x & 255;
  int t = threadIdx.x;
  int lane = t & 31, w = t >> 5;
  int g = lane >> 4, mr = lane & 15, col = lane & 15;

  // ---- Stage 0: stage 96KB of bf16 weights into LDS (16B per lane per iter) ----
  {
    const char* gw = (const char*)Wem;
#if defined(HAVE_ASYNC_LDS)
#pragma unroll 4
    for (int r = 0; r < 24; ++r) {
      int idx = (t + (r << 8)) << 4;
      __builtin_amdgcn_global_load_async_to_lds_b128(
          (gptr4)(gw + idx), (lptr4)(void*)(smem + idx), 0, 0);
    }
    __builtin_amdgcn_s_wait_asynccnt(0);
#else
    const uint4* g4 = (const uint4*)gw;
    uint4* s4 = (uint4*)smem;
#pragma unroll 4
    for (int r = 0; r < 24; ++r) {
      int idx = t + (r << 8);
      s4[idx] = g4[idx];
    }
#endif
  }
  qrow[t] = QKV[(size_t)(b * NN + i) * 768 + t];
  ye1s[t] = yev[b * NN + t];
  ye2s[t] = yev[512 + b * NN + t];
  float xmi = mask[b * NN + i] ? 1.f : 0.f;
  __syncthreads();

  // ---- Stage 1: build Y slab ----
  for (int mt = w; mt < 16; mt += 8) {
    int ja = mt * 16 + mr;
    const float* Erow = E + ((size_t)(b * NN + i) * NN + ja) * DE;
    v16bf a0 = load_a_frag_f32(Erow, g);
    v16bf a1 = load_a_frag_f32(Erow + 32, g);
    for (int nt = 0; nt < 16; ++nt) {
      int cb = nt * 16 + col;
      v16bf bm0, bm1, ba0, ba1;
#pragma unroll
      for (int e = 0; e < 16; ++e) {
        int k0 = bmap(e, g);
        bm0[e] = sWem[k0 * DX + cb];
        bm1[e] = sWem[(32 + k0) * DX + cb];
        ba0[e] = sWea[k0 * DX + cb];
        ba1[e] = sWea[(32 + k0) * DX + cb];
      }
      v8f acc1 = vzero8(), acc2 = vzero8();
      acc1 = wmma_bf16(a0, bm0, acc1);
      acc1 = wmma_bf16(a1, bm1, acc1);
      acc2 = wmma_bf16(a0, ba0, acc2);
      acc2 = wmma_bf16(a1, ba1, acc2);
      float q = qrow[cb];
#pragma unroll
      for (int v = 0; v < 8; ++v) {
        int j = mt * 16 + v + 8 * g;
        float xmj = mask[b * NN + j] ? 1.f : 0.f;
        float mm = xmi * xmj;
        float e1 = (acc1[v] + emul_b[cb]) * mm;
        float e2 = (acc2[v] + eadd_b[cb]) * mm;
        float kv = QKV[(size_t)(b * NN + j) * 768 + 256 + cb];
        float yv = q * kv * RS * (e1 + 1.f) + e2;
        Ybf[j * DX + cb] = (bf16)yv;
      }
    }
  }
  __syncthreads();

  // ---- Stage 2: e_out GEMM (M=256 j, K=256 c, N=64) ----
  for (int tt = w; tt < 64; tt += 8) {
    int mt = tt >> 2, nt = tt & 3;
    v8f acc = vzero8();
    int m = mt * 16 + mr;
    for (int kk = 0; kk < 256; kk += 32) {
      v16bf a, bb;
#pragma unroll
      for (int e = 0; e < 16; ++e) {
        int ka = kk + amap(e, g);
        a[e] = (bf16)(ye1s[ka] + (ye2s[ka] + 1.f) * (float)Ybf[m * DX + ka]);
        int kb2 = kk + bmap(e, g);
        bb[e] = sWeo[kb2 * DE + nt * 16 + col];
      }
      acc = wmma_bf16(a, bb, acc);
    }
#pragma unroll
    for (int v = 0; v < 8; ++v) {
      int j = mt * 16 + v + 8 * g;
      int ce = nt * 16 + col;
      float xmj = mask[b * NN + j] ? 1.f : 0.f;
      outE[((size_t)(b * NN + i) * NN + j) * DE + ce] = (acc[v] + eout_b[ce]) * xmi * xmj;
    }
  }

  // ---- Stage 3: per-channel softmax over j + weighted V ----
  {
    float mx = -__builtin_inff();
    for (int j = 0; j < NN; ++j) {
      if (mask[b * NN + j]) {
        float yv = (float)Ybf[j * DX + t];
        mx = fmaxf(mx, yv);
      }
    }
    float s = 0.f, wvacc = 0.f;
    for (int j = 0; j < NN; ++j) {
      if (mask[b * NN + j]) {
        float e = __expf((float)Ybf[j * DX + t] - mx);
        s += e;
        wvacc += e * QKV[(size_t)(b * NN + j) * 768 + 512 + t];
      }
    }
    float wvv = wvacc / s;
    tmpv[t] = yev[1024 + b * NN + t] + (yev[1536 + b * NN + t] + 1.f) * wvv;
  }
  __syncthreads();

  // ---- Stage 4: newX row ----
  {
    float acc = x_out_b[t];
    for (int k = 0; k < DX; ++k) acc += tmpv[k] * x_out_w[k * DX + t];
    outX[(size_t)(b * NN + i) * DX + t] = acc * xmi;
  }
}

// ------------------------------------------------------------------
// Kernel 5: X statistics -> zx (bs, 1024) = [mean | min | max | std]
// ------------------------------------------------------------------
__global__ __launch_bounds__(256) void k_xstats(const float* __restrict__ X, float* __restrict__ zx) {
  __shared__ float ssum[256], smin[256], smax[256], ssq[256];
  int b = blockIdx.x >> 8, c = blockIdx.x & 255, t = threadIdx.x;
  float v = X[(size_t)(b * NN + t) * DX + c];
  ssum[t] = v; smin[t] = v; smax[t] = v; ssq[t] = v * v;
  __syncthreads();
  for (int off = 128; off > 0; off >>= 1) {
    if (t < off) {
      ssum[t] += ssum[t + off];
      smin[t] = fminf(smin[t], smin[t + off]);
      smax[t] = fmaxf(smax[t], smax[t + off]);
      ssq[t] += ssq[t + off];
    }
    __syncthreads();
  }
  if (t == 0) {
    float mean = ssum[0] / 256.f;
    float var = fmaxf(ssq[0] - 256.f * mean * mean, 0.f) / 255.f;
    zx[b * 1024 + c]       = mean;
    zx[b * 1024 + 256 + c] = smin[0];
    zx[b * 1024 + 512 + c] = smax[0];
    zx[b * 1024 + 768 + c] = sqrtf(var);
  }
}

// ------------------------------------------------------------------
// Kernel 6: E statistics -> ze (bs, 256) = [mean | min | max | std]
// ------------------------------------------------------------------
__global__ __launch_bounds__(256) void k_estats(const float* __restrict__ E, float* __restrict__ ze) {
  __shared__ float ssum[256], smin[256], smax[256], ssq[256];
  int b = blockIdx.x >> 6, ce = blockIdx.x & 63, t = threadIdx.x;
  float s = 0.f, sq = 0.f, mn = __builtin_inff(), mx = -__builtin_inff();
  for (int r = 0; r < 256; ++r) {
    int p = t + (r << 8);
    float v = E[((size_t)b * 65536 + p) * DE + ce];
    s += v; sq += v * v;
    mn = fminf(mn, v); mx = fmaxf(mx, v);
  }
  ssum[t] = s; smin[t] = mn; smax[t] = mx; ssq[t] = sq;
  __syncthreads();
  for (int off = 128; off > 0; off >>= 1) {
    if (t < off) {
      ssum[t] += ssum[t + off];
      smin[t] = fminf(smin[t], smin[t + off]);
      smax[t] = fmaxf(smax[t], smax[t + off]);
      ssq[t] += ssq[t + off];
    }
    __syncthreads();
  }
  if (t == 0) {
    float cnt = 65536.f;
    float mean = ssum[0] / cnt;
    float var = fmaxf(ssq[0] - cnt * mean * mean, 0.f) / (cnt - 1.f);
    ze[b * 256 + ce]       = mean;
    ze[b * 256 + 64 + ce]  = smin[0];
    ze[b * 256 + 128 + ce] = smax[0];
    ze[b * 256 + 192 + ce] = sqrtf(var);
  }
}

// ------------------------------------------------------------------
// Kernel 7: final y MLP. Single block; 128 active threads ((b,c)).
// ------------------------------------------------------------------
__global__ __launch_bounds__(256) void k_ynew(
    const float* __restrict__ y, const float* __restrict__ zx, const float* __restrict__ ze,
    const float* __restrict__ yyw, const float* __restrict__ yyb,
    const float* __restrict__ xyw, const float* __restrict__ xyb,
    const float* __restrict__ eyw, const float* __restrict__ eyb,
    const float* __restrict__ o1w, const float* __restrict__ o1b,
    const float* __restrict__ o2w, const float* __restrict__ o2b,
    float* __restrict__ outy) {
  __shared__ float t1[128], hb[128];
  int t = threadIdx.x;
  int b = (t >> 6) & 1, c = t & 63;
  if (t < 128) {
    float acc = yyb[c] + xyb[c] + eyb[c];
    for (int k = 0; k < 64; ++k)   acc += y[b * 64 + k] * yyw[k * 64 + c];
    for (int k = 0; k < 1024; ++k) acc += zx[b * 1024 + k] * xyw[k * 64 + c];
    for (int k = 0; k < 256; ++k)  acc += ze[b * 256 + k] * eyw[k * 64 + c];
    t1[t] = acc;
  }
  __syncthreads();
  if (t < 128) {
    float acc = o1b[c];
    for (int k = 0; k < 64; ++k) acc += t1[b * 64 + k] * o1w[k * 64 + c];
    hb[t] = fmaxf(acc, 0.f);
  }
  __syncthreads();
  if (t < 128) {
    float acc = o2b[c];
    for (int k = 0; k < 64; ++k) acc += hb[b * 64 + k] * o2w[k * 64 + c];
    outy[t] = acc;
  }
}

// ------------------------------------------------------------------
extern "C" void kernel_launch(void* const* d_in, const int* in_sizes, int n_in,
                              void* d_out, int out_size, void* d_ws, size_t ws_size,
                              hipStream_t stream) {
  (void)in_sizes; (void)n_in; (void)out_size; (void)ws_size;

  const float* X = (const float*)d_in[0];
  const float* E = (const float*)d_in[1];
  const float* y = (const float*)d_in[2];
  const unsigned char* mask = (const unsigned char*)d_in[3];
  const float* q_w = (const float*)d_in[4];   const float* q_b = (const float*)d_in[5];
  const float* k_w = (const float*)d_in[6];   const float* k_b = (const float*)d_in[7];
  const float* v_w = (const float*)d_in[8];   const float* v_b = (const float*)d_in[9];
  const float* e_mul_w = (const float*)d_in[10]; const float* e_mul_b = (const float*)d_in[11];
  const float* e_add_w = (const float*)d_in[12]; const float* e_add_b = (const float*)d_in[13];
  const float* y_e_mul_w = (const float*)d_in[14]; const float* y_e_mul_b = (const float*)d_in[15];
  const float* y_e_add_w = (const float*)d_in[16]; const float* y_e_add_b = (const float*)d_in[17];
  const float* y_x_mul_w = (const float*)d_in[18]; const float* y_x_mul_b = (const float*)d_in[19];
  const float* y_x_add_w = (const float*)d_in[20]; const float* y_x_add_b = (const float*)d_in[21];
  const float* y_y_w = (const float*)d_in[22]; const float* y_y_b = (const float*)d_in[23];
  const float* x_y_w = (const float*)d_in[24]; const float* x_y_b = (const float*)d_in[25];
  const float* e_y_w = (const float*)d_in[26]; const float* e_y_b = (const float*)d_in[27];
  const float* x_out_w = (const float*)d_in[28]; const float* x_out_b = (const float*)d_in[29];
  const float* e_out_w = (const float*)d_in[30]; const float* e_out_b = (const float*)d_in[31];
  const float* y_out1_w = (const float*)d_in[32]; const float* y_out1_b = (const float*)d_in[33];
  const float* y_out2_w = (const float*)d_in[34]; const float* y_out2_b = (const float*)d_in[35];

  // workspace layout (bytes); Wem|Wea|Weo deliberately contiguous (96KB)
  char* ws = (char*)d_ws;
  bf16*  Wqkv = (bf16*)(ws + 0);         // 256*768 bf16   = 393216 B
  bf16*  Wem  = (bf16*)(ws + 393216);    // 64*256 bf16    = 32768 B
  bf16*  Wea  = (bf16*)(ws + 425984);    // 64*256 bf16    = 32768 B
  bf16*  Weo  = (bf16*)(ws + 458752);    // 256*64 bf16    = 32768 B
  float* QKV  = (float*)(ws + 491520);   // 512*768 f32    = 1572864 B
  float* yev  = (float*)(ws + 2064384);  // 2048 f32       = 8192 B
  float* zx   = (float*)(ws + 2072576);  // 2048 f32       = 8192 B
  float* ze   = (float*)(ws + 2080768);  // 512 f32        = 2048 B

  float* outX = (float*)d_out;                     // 2*256*256
  float* outE = outX + 131072;                     // 2*256*256*64
  float* outY = outX + 131072 + 8388608;           // 2*64

  k_cvt_weights<<<960, 256, 0, stream>>>(q_w, k_w, v_w, e_mul_w, e_add_w, e_out_w,
                                         Wqkv, Wem, Wea, Weo);
  k_yvec<<<8, 256, 0, stream>>>(y, y_e_add_w, y_e_add_b, y_e_mul_w, y_e_mul_b,
                                y_x_add_w, y_x_add_b, y_x_mul_w, y_x_mul_b, yev);
  k_qkv_gemm<<<dim3(4, 12), 256, 0, stream>>>(X, Wqkv, q_b, k_b, v_b, mask, QKV);

  k_fused<<<512, 256, FUSED_LDS_BYTES, stream>>>(E, mask, QKV, Wem,
                                                 e_mul_b, e_add_b, e_out_b, yev,
                                                 x_out_w, x_out_b, outX, outE);

  k_xstats<<<512, 256, 0, stream>>>(X, zx);
  k_estats<<<128, 256, 0, stream>>>(E, ze);
  k_ynew<<<1, 256, 0, stream>>>(y, zx, ze, y_y_w, y_y_b, x_y_w, x_y_b, e_y_w, e_y_b,
                                y_out1_w, y_out1_b, y_out2_w, y_out2_b, outY);
}
